// PointnetSAModule_76854144795252
// MI455X (gfx1250) — matrix operand order repaired and checked by reading, not compiled, not run on hardware
//
#include <hip/hip_runtime.h>
#include <hip/hip_bf16.h>
#include <math.h>

#define Bn      8
#define Nn      8192
#define NPOINT  1024
#define NSAMPLE 32
#define R2      0.04f    // 0.2^2
#define EPSV    1e-5f

typedef __attribute__((ext_vector_type(16))) _Float16 v16h;
typedef __attribute__((ext_vector_type(8)))  _Float16 v8h;
typedef __attribute__((ext_vector_type(8)))  float    v8f;
typedef __attribute__((ext_vector_type(4)))  float    v4f;

// ---------------------------------------------------------------------------
// Kernel 1: furthest point sampling. One workgroup per batch.
// Each thread keeps its 8 points' xyz and running min-distances in REGISTERS;
// the inner loop touches no global memory and no LDS data arrays — only the
// 32-wave argmax tree + broadcast. Winner writes curPt/new_xyz from registers.
// ---------------------------------------------------------------------------
__global__ __launch_bounds__(1024) void fps_kernel(const float* __restrict__ xyz,
                                                   float* __restrict__ new_xyz) {
    __shared__ float redV[32];
    __shared__ int   redI[32];
    __shared__ float curPt[3];
    __shared__ int   curIdx;

    const int b    = blockIdx.x;
    const int tid  = threadIdx.x;
    const int lane = tid & 31;
    const int wid  = tid >> 5;
    const float* xb = xyz + (size_t)b * Nn * 3;

    float xr[8], yr[8], zr[8], dr[8];
    #pragma unroll
    for (int k = 0; k < 8; ++k) {
        int j = tid + k * 1024;
        xr[k] = xb[j * 3 + 0];
        yr[k] = xb[j * 3 + 1];
        zr[k] = xb[j * 3 + 2];
        dr[k] = 3.4e38f;
    }
    if (tid == 0) {
        curIdx = 0;
        curPt[0] = xr[0]; curPt[1] = yr[0]; curPt[2] = zr[0];
        float* o = new_xyz + (size_t)b * NPOINT * 3;
        o[0] = xr[0]; o[1] = yr[0]; o[2] = zr[0];
    }
    __syncthreads();

    for (int it = 0; it < NPOINT; ++it) {
        const float cx = curPt[0], cy = curPt[1], cz = curPt[2];

        float bv = -1.0f; int bi = 0;
        #pragma unroll
        for (int k = 0; k < 8; ++k) {
            float dx = xr[k] - cx, dy = yr[k] - cy, dz = zr[k] - cz;
            float d  = dx * dx + dy * dy + dz * dz;
            float nd = fminf(dr[k], d);
            dr[k] = nd;
            if (nd > bv) { bv = nd; bi = tid + k * 1024; }
        }
        // wave32 argmax, first-index tie break
        #pragma unroll
        for (int off = 16; off > 0; off >>= 1) {
            float ov = __shfl_down(bv, off, 32);
            int   oi = __shfl_down(bi, off, 32);
            if (ov > bv || (ov == bv && oi < bi)) { bv = ov; bi = oi; }
        }
        if (lane == 0) { redV[wid] = bv; redI[wid] = bi; }
        __syncthreads();
        if (wid == 0) {
            float v = redV[lane]; int i = redI[lane];
            #pragma unroll
            for (int off = 16; off > 0; off >>= 1) {
                float ov = __shfl_down(v, off, 32);
                int   oi = __shfl_down(i, off, 32);
                if (ov > v || (ov == v && oi < i)) { v = ov; i = oi; }
            }
            if (lane == 0) curIdx = i;
        }
        __syncthreads();
        // winning thread broadcasts its point from registers (no global load)
        int ci = curIdx;
        if (tid == (ci & 1023)) {
            int k = ci >> 10;
            curPt[0] = xr[k]; curPt[1] = yr[k]; curPt[2] = zr[k];
            if (it + 1 < NPOINT) {
                float* o = new_xyz + ((size_t)b * NPOINT + it + 1) * 3;
                o[0] = xr[k]; o[1] = yr[k]; o[2] = zr[k];
            }
        }
        __syncthreads();
    }
}

// ---------------------------------------------------------------------------
// Kernel 2: fused ball-query + 3-layer MLP + maxpool, one wave per center.
// Transposed GEMM formulation D' = W x H^T so that:
//   A operand = weights, pre-packed into exact A-fragment layout in LDS
//   B operand = activations, per-lane slice is CONTIGUOUS -> v16h LDS loads
//   D tiles give 8 consecutive channels per lane -> v8h contiguous stores
// ---------------------------------------------------------------------------
__global__ __launch_bounds__(256) void group_mlp_kernel(
        const float* __restrict__ xyz, const float* __restrict__ new_xyz,
        const float* __restrict__ w1, const float* __restrict__ b1,
        const float* __restrict__ g1, const float* __restrict__ be1,
        const float* __restrict__ m1, const float* __restrict__ v1,
        const float* __restrict__ w2, const float* __restrict__ b2,
        const float* __restrict__ g2, const float* __restrict__ be2,
        const float* __restrict__ m2, const float* __restrict__ v2,
        const float* __restrict__ w3, const float* __restrict__ b3,
        const float* __restrict__ g3, const float* __restrict__ be3,
        const float* __restrict__ m3, const float* __restrict__ v3,
        float* __restrict__ feat) {

    // pre-packed A fragments: [mtile][ktile][lane][16 halves]
    __shared__ __align__(32) _Float16 w2p[4][2][32][16];    //  8 KB
    __shared__ __align__(32) _Float16 w3p[8][2][32][16];    // 16 KB
    __shared__ __align__(32) _Float16 hbuf[8][32][64];      // 32 KB  [point][ch]
    __shared__ int   ballIdx[8][32];
    __shared__ __align__(16) float w1s[64 * 3];
    __shared__ __align__(16) float a1s[64], c1s[64];
    __shared__ __align__(16) float a2s[64], c2s[64];
    __shared__ __align__(16) float a3s[128], c3s[128];

    const int tid  = threadIdx.x;
    const int lane = tid & 31;
    const int w    = tid >> 5;
    const int col  = lane & 15;
    const int hi   = lane >> 4;

    // ---- stage folded BN constants + pre-packed f16 weight fragments ----
    for (int i = tid; i < 64; i += 256) {
        float a = g1[i] * rsqrtf(v1[i] + EPSV);
        a1s[i] = a; c1s[i] = a * (b1[i] - m1[i]) + be1[i];
        float a2 = g2[i] * rsqrtf(v2[i] + EPSV);
        a2s[i] = a2; c2s[i] = a2 * (b2[i] - m2[i]) + be2[i];
    }
    for (int i = tid; i < 128; i += 256) {
        float a = g3[i] * rsqrtf(v3[i] + EPSV);
        a3s[i] = a; c3s[i] = a * (b3[i] - m3[i]) + be3[i];
    }
    for (int i = tid; i < 192; i += 256) w1s[i] = w1[i];
    // A-fragment packing (contiguous k order, same as B loads): lane row = lane&15,
    // half h -> k = kt*32 + h + (lane>=16 ? 16 : 0)
    for (int i = tid; i < 4 * 2 * 32 * 16; i += 256) {
        int h  = i & 15, ln = (i >> 4) & 31, kt = (i >> 9) & 1, mt = (i >> 10) & 3;
        int cl = ln & 15, hh = ln >> 4;
        int k  = kt * 32 + h + hh * 16;
        w2p[mt][kt][ln][h] = (_Float16)w2[(mt * 16 + cl) * 64 + k];
    }
    for (int i = tid; i < 8 * 2 * 32 * 16; i += 256) {
        int h  = i & 15, ln = (i >> 4) & 31, kt = (i >> 9) & 1, mt = (i >> 10) & 7;
        int cl = ln & 15, hh = ln >> 4;
        int k  = kt * 32 + h + hh * 16;
        w3p[mt][kt][ln][h] = (_Float16)w3[(mt * 16 + cl) * 64 + k];
    }
    __syncthreads();

    const int center = blockIdx.x * 8 + w;                 // 0..8191
    const int b = center / NPOINT;
    const int s = center % NPOINT;
    const float* xb = xyz + (size_t)b * Nn * 3;

    // ---- ball query: first NSAMPLE indices within radius, in index order ----
    const float cx = new_xyz[(size_t)center * 3 + 0];
    const float cy = new_xyz[(size_t)center * 3 + 1];
    const float cz = new_xyz[(size_t)center * 3 + 2];
    ballIdx[w][lane] = 0;                                  // pad -> index 0
    int cnt = 0;
    for (int base = 0; base < Nn && cnt < NSAMPLE; base += 32) {
        int j = base + lane;
        float dx = xb[j * 3 + 0] - cx;
        float dy = xb[j * 3 + 1] - cy;
        float dz = xb[j * 3 + 2] - cz;
        bool valid = (dx * dx + dy * dy + dz * dz) < R2;
        unsigned mask = (unsigned)__ballot(valid);
        if (valid) {
            int pos = cnt + __popc(mask & ((1u << lane) - 1u));
            if (pos < NSAMPLE) ballIdx[w][pos] = j;
        }
        cnt += __popc(mask);
    }
    const int pidx = ballIdx[w][lane];

    // ---- layer 1 (K=3, VALU): lane = sample point; vectorized v16h stores ----
    {
        float px = xb[pidx * 3 + 0], py = xb[pidx * 3 + 1], pz = xb[pidx * 3 + 2];
        #pragma unroll
        for (int q = 0; q < 4; ++q) {
            v16h t;
            #pragma unroll
            for (int cc = 0; cc < 16; ++cc) {
                int c = q * 16 + cc;
                float d = px * w1s[c * 3 + 0] + py * w1s[c * 3 + 1] + pz * w1s[c * 3 + 2];
                t[cc] = (_Float16)fmaxf(a1s[c] * d + c1s[c], 0.0f);
            }
            *(v16h*)&hbuf[w][lane][q * 16] = t;
        }
    }

    // ---- layer 2: D' = W2 x H1^T ; M'=64(oc) 4 tiles, N'=32(pts) 2 tiles ----
    // B (activation) fragment: lane col -> h[point=ntile*16+col][kt*32+hi*16 ..+15]
    v16h bact[2][2];
    #pragma unroll
    for (int nt = 0; nt < 2; ++nt)
        #pragma unroll
        for (int kt = 0; kt < 2; ++kt)
            bact[nt][kt] = *(const v16h*)&hbuf[w][nt * 16 + col][kt * 32 + hi * 16];

    #pragma unroll
    for (int mt = 0; mt < 4; ++mt) {
        v16h wa0 = *(const v16h*)&w2p[mt][0][lane][0];
        v16h wa1 = *(const v16h*)&w2p[mt][1][lane][0];
        const int ocb = mt * 16 + hi * 8;                  // 8 consecutive channels
        v4f av0 = *(const v4f*)&a2s[ocb], av1 = *(const v4f*)&a2s[ocb + 4];
        v4f cv0 = *(const v4f*)&c2s[ocb], cv1 = *(const v4f*)&c2s[ocb + 4];
        #pragma unroll
        for (int nt = 0; nt < 2; ++nt) {
            v8f acc = {};
            acc = __builtin_amdgcn_wmma_f32_16x16x32_f16(false, wa0, false, bact[nt][0],
                                                         (short)0, acc, false, false);
            acc = __builtin_amdgcn_wmma_f32_16x16x32_f16(false, wa1, false, bact[nt][1],
                                                         (short)0, acc, false, false);
            v8h outh;
            #pragma unroll
            for (int r = 0; r < 8; ++r) {
                float a = (r < 4) ? av0[r] : av1[r - 4];
                float c = (r < 4) ? cv0[r] : cv1[r - 4];
                outh[r] = (_Float16)fmaxf(acc[r] * a + c, 0.0f);
            }
            *(v8h*)&hbuf[w][nt * 16 + col][ocb] = outh;    // contiguous 16B store
        }
    }

    // ---- layer 3: D' = W3 x H2^T ; M'=128(oc) 8 tiles; fused maxpool ----
    v16h bact3[2][2];
    #pragma unroll
    for (int nt = 0; nt < 2; ++nt)
        #pragma unroll
        for (int kt = 0; kt < 2; ++kt)
            bact3[nt][kt] = *(const v16h*)&hbuf[w][nt * 16 + col][kt * 32 + hi * 16];

    for (int mt = 0; mt < 8; ++mt) {
        v16h wa0 = *(const v16h*)&w3p[mt][0][lane][0];
        v16h wa1 = *(const v16h*)&w3p[mt][1][lane][0];
        const int ocb = mt * 16 + hi * 8;
        v4f av0 = *(const v4f*)&a3s[ocb], av1 = *(const v4f*)&a3s[ocb + 4];
        v4f cv0 = *(const v4f*)&c3s[ocb], cv1 = *(const v4f*)&c3s[ocb + 4];

        v8f acc0 = {}, acc1 = {};
        acc0 = __builtin_amdgcn_wmma_f32_16x16x32_f16(false, wa0, false, bact3[0][0],
                                                      (short)0, acc0, false, false);
        acc0 = __builtin_amdgcn_wmma_f32_16x16x32_f16(false, wa1, false, bact3[0][1],
                                                      (short)0, acc0, false, false);
        acc1 = __builtin_amdgcn_wmma_f32_16x16x32_f16(false, wa0, false, bact3[1][0],
                                                      (short)0, acc1, false, false);
        acc1 = __builtin_amdgcn_wmma_f32_16x16x32_f16(false, wa1, false, bact3[1][1],
                                                      (short)0, acc1, false, false);

        float mr[8];
        #pragma unroll
        for (int r = 0; r < 8; ++r) {
            float a = (r < 4) ? av0[r] : av1[r - 4];
            float c = (r < 4) ? cv0[r] : cv1[r - 4];
            float u0 = fmaxf(acc0[r] * a + c, 0.0f);       // 16 points of tile 0
            float u1 = fmaxf(acc1[r] * a + c, 0.0f);       // 16 points of tile 1
            float m  = fmaxf(u0, u1);
            // max over the 16 points held across the 16-lane half
            m = fmaxf(m, __shfl_xor(m, 1, 32));
            m = fmaxf(m, __shfl_xor(m, 2, 32));
            m = fmaxf(m, __shfl_xor(m, 4, 32));
            m = fmaxf(m, __shfl_xor(m, 8, 32));
            mr[r] = m;
        }
        // lanes col<8 each store one of the 8 channels of this half
        float outv = mr[0];
        #pragma unroll
        for (int r = 1; r < 8; ++r) outv = (col == r) ? mr[r] : outv;
        if (col < 8)
            feat[((size_t)b * 128 + (ocb + col)) * NPOINT + s] = outv;
    }
}

// ---------------------------------------------------------------------------
extern "C" void kernel_launch(void* const* d_in, const int* in_sizes, int n_in,
                              void* d_out, int out_size, void* d_ws, size_t ws_size,
                              hipStream_t stream) {
    (void)in_sizes; (void)n_in; (void)d_ws; (void)ws_size; (void)out_size;
    const float* xyz = (const float*)d_in[0];
    const float* w1  = (const float*)d_in[1];
    const float* b1  = (const float*)d_in[2];
    const float* g1  = (const float*)d_in[3];
    const float* be1 = (const float*)d_in[4];
    const float* m1  = (const float*)d_in[5];
    const float* v1  = (const float*)d_in[6];
    const float* w2  = (const float*)d_in[7];
    const float* b2  = (const float*)d_in[8];
    const float* g2  = (const float*)d_in[9];
    const float* be2 = (const float*)d_in[10];
    const float* m2  = (const float*)d_in[11];
    const float* v2  = (const float*)d_in[12];
    const float* w3  = (const float*)d_in[13];
    const float* b3  = (const float*)d_in[14];
    const float* g3  = (const float*)d_in[15];
    const float* be3 = (const float*)d_in[16];
    const float* m3  = (const float*)d_in[17];
    const float* v3  = (const float*)d_in[18];

    float* out     = (float*)d_out;
    float* new_xyz = out;                                  // B*NPOINT*3
    float* feat    = out + (size_t)Bn * NPOINT * 3;        // B*128*NPOINT

    fps_kernel<<<Bn, 1024, 0, stream>>>(xyz, new_xyz);

    group_mlp_kernel<<<(Bn * NPOINT) / 8, 256, 0, stream>>>(
        xyz, new_xyz,
        w1, b1, g1, be1, m1, v1,
        w2, b2, g2, be2, m2, v2,
        w3, b3, g3, be3, m3, v3,
        feat);
}
